// ParetoStatePredictorMIS_47614007443927
// MI455X (gfx1250) — matrix operand order repaired and checked by reading, not compiled, not run on hardware
//
#include <hip/hip_runtime.h>
#include <hip/hip_bf16.h>
#include <math.h>

// ---------------------------------------------------------------------------
// ParetoStatePredictorMIS for MI455X (gfx1250, wave32, WMMA).
//
// The edge tensor path (B*N*N*D = 128MB, ~34 GFLOP of edge MLPs/LNs) is DEAD
// w.r.t. the output -- only the node path reaches the heads. We skip it.
// All live GEMMs run on v_wmma_f32_16x16x32_f16 with f16 operands staged in
// LDS and fp32 accumulation. Tile loaders are branch-free (all shapes are
// tile-aligned) and vectorized (b128 global loads). The Gram-matrix GEMM
// (G = E^T E for the SVD features) reads a pre-converted f16 copy of E and
// fills its LDS tiles with gfx1250 async global->LDS copies (ASYNCcnt path).
// ---------------------------------------------------------------------------

typedef _Float16 v16h __attribute__((ext_vector_type(16)));
typedef _Float16 h4   __attribute__((ext_vector_type(4)));
typedef float    v8f  __attribute__((ext_vector_type(8)));
typedef int      v4i  __attribute__((ext_vector_type(4)));

#define WMMA_F16(a, b, c) \
    __builtin_amdgcn_wmma_f32_16x16x32_f16(false, (a), false, (b), (short)0, (c), false, false)

#if defined(__HIP_DEVICE_COMPILE__) && defined(__has_builtin)
#if __has_builtin(__builtin_amdgcn_global_load_async_to_lds_b128) && \
    __has_builtin(__builtin_amdgcn_s_wait_asynccnt)
#define HAVE_ASYNC_LDS 1
#endif
#endif
#ifndef HAVE_ASYNC_LDS
#define HAVE_ASYNC_LDS 0
#endif

#if HAVE_ASYNC_LDS
typedef __attribute__((address_space(1))) v4i* gvec_ptr;
typedef __attribute__((address_space(3))) v4i* lvec_ptr;
#endif

// A fragment (16x32 f16), tile row-major [16][32] halfs in LDS.
// Lanes 0-15 (m=lane): K 0..15; lanes 16-31: K 16..31.
__device__ __forceinline__ v16h frag_ld_a(const _Float16* tile) {
    int lane = threadIdx.x & 31;
    int r = lane & 15;
    int kb = (lane & 16) ? 16 : 0;
    const _Float16* p = tile + r * 32;
    union { v16h v; float4 q[2]; } u;
    u.q[0] = *(const float4*)(p + kb);
    u.q[1] = *(const float4*)(p + kb + 8);
    return u.v;
}

// B fragment (32x16 f16), tile stored n-major [16 n][32 k] halfs in LDS.
// Lanes 0-15 (n=lane): K=0..15; lanes 16-31: K=16..31.
__device__ __forceinline__ v16h frag_ld_b(const _Float16* tile) {
    int lane = threadIdx.x & 31;
    int nloc = lane & 15;
    int kb = (lane & 16) ? 16 : 0;
    const _Float16* p = tile + nloc * 32 + kb;
    union { v16h v; float4 q[2]; } u;
    u.q[0] = *(const float4*)(p);
    u.q[1] = *(const float4*)(p + 8);
    return u.v;
}

// Fragment loads from k-major tiles Ek[k][64] (used by the Gram kernel).
__device__ __forceinline__ v16h frag_ld_a_kmaj(const _Float16* Ek, int mloc) {
    int lane = threadIdx.x & 31;
    int m = mloc + (lane & 15);
    int kb = (lane & 16) ? 16 : 0;
    v16h f;
#pragma unroll
    for (int i = 0; i < 8; ++i) {
        f[i] = Ek[(kb + i) * 64 + m];
        f[i + 8] = Ek[(kb + 8 + i) * 64 + m];
    }
    return f;
}
__device__ __forceinline__ v16h frag_ld_b_kmaj(const _Float16* Ek, int nloc) {
    int lane = threadIdx.x & 31;
    int n = nloc + (lane & 15);
    int kb = (lane & 16) ? 16 : 0;
    v16h f;
#pragma unroll
    for (int i = 0; i < 16; ++i) f[i] = Ek[(kb + i) * 64 + n];
    return f;
}

// ---------------------------------------------------------------------------
// Row-major x row-major WMMA GEMM, tile-aligned shapes only:
// M%64==0, N%64==0, K%32==0. C = [relu](A@B + bias) + resid.
// WG tile 64x64, 4 waves (2x2), each wave 2x2 16x16 frags, K step 32.
// Branch-free vectorized loader: 8 x global b128 per thread-tile.
// ---------------------------------------------------------------------------
__global__ __launch_bounds__(128)
void wmma_gemm_rr(const float* __restrict__ A, int lda,
                  const float* __restrict__ Bm, int ldb,
                  float* __restrict__ C, int ldc,
                  const float* __restrict__ bias, const float* __restrict__ resid,
                  int relu, int K) {
    __shared__ __align__(16) _Float16 As[64 * 32];  // [m][k]
    __shared__ __align__(16) _Float16 Bs[64 * 32];  // n-major [n][k]
    int tid = threadIdx.x;
    int lane = tid & 31;
    int wave = tid >> 5;
    int m_base = blockIdx.y * 64;
    int n_base = blockIdx.x * 64;
    int wm = wave >> 1, wn = wave & 1;
    v8f acc00 = {}, acc01 = {}, acc10 = {}, acc11 = {};
    int ksteps = K >> 5;
    for (int kt = 0; kt < ksteps; ++kt) {
        int k0 = kt << 5;
        if (tid == 0 && kt + 1 < ksteps)
            __builtin_prefetch(A + (long)m_base * lda + k0 + 32, 0, 0);
#pragma unroll
        for (int j = 0; j < 4; ++j) {
            int f = tid + (j << 7);  // 0..511 float4 slots
            {   // A tile: 64 rows x 8 float4
                int row = f >> 3, seg = (f & 7) << 2;
                float4 av = *(const float4*)(A + (long)(m_base + row) * lda + k0 + seg);
                h4 hv;
                hv[0] = (_Float16)av.x; hv[1] = (_Float16)av.y;
                hv[2] = (_Float16)av.z; hv[3] = (_Float16)av.w;
                *(h4*)(As + row * 32 + seg) = hv;
            }
            {   // B tile: 32 k-rows x 16 float4 along n, scatter to n-major
                int kk = f >> 4, seg = (f & 15) << 2;
                float4 bv = *(const float4*)(Bm + (long)(k0 + kk) * ldb + n_base + seg);
                Bs[(seg + 0) * 32 + kk] = (_Float16)bv.x;
                Bs[(seg + 1) * 32 + kk] = (_Float16)bv.y;
                Bs[(seg + 2) * 32 + kk] = (_Float16)bv.z;
                Bs[(seg + 3) * 32 + kk] = (_Float16)bv.w;
            }
        }
        __syncthreads();
        v16h a0 = frag_ld_a(As + (wm * 32) * 32);
        v16h a1 = frag_ld_a(As + (wm * 32 + 16) * 32);
        v16h b0 = frag_ld_b(Bs + (wn * 32) * 32);
        v16h b1 = frag_ld_b(Bs + (wn * 32 + 16) * 32);
        acc00 = WMMA_F16(a0, b0, acc00);
        acc01 = WMMA_F16(a0, b1, acc01);
        acc10 = WMMA_F16(a1, b0, acc10);
        acc11 = WMMA_F16(a1, b1, acc11);
        __syncthreads();
    }
    int rowoff = (lane & 16) ? 8 : 0;
    int col_l = lane & 15;
    v8f* accs[2][2] = {{&acc00, &acc01}, {&acc10, &acc11}};
    for (int mi = 0; mi < 2; ++mi)
        for (int ni = 0; ni < 2; ++ni) {
            int gcol = n_base + wn * 32 + ni * 16 + col_l;
#pragma unroll
            for (int r = 0; r < 8; ++r) {
                long grow = m_base + wm * 32 + mi * 16 + r + rowoff;
                float vv = (*accs[mi][ni])[r];
                if (bias) vv += bias[gcol];
                if (relu) vv = fmaxf(vv, 0.f);
                if (resid) vv += resid[grow * ldc + gcol];
                C[grow * ldc + gcol] = vv;
            }
        }
}

// ---------------------------------------------------------------------------
// G[b] = E[b]^T E[b], 512x512, from an f16 copy of E. Both the A-window
// (cols m_base..+63 of E rows) and B-window tiles are contiguous 16-byte row
// segments -> filled with async global->LDS b128 copies (ASYNCcnt).
// ---------------------------------------------------------------------------
__global__ __launch_bounds__(128)
void wmma_gram(const _Float16* __restrict__ Eh, float* __restrict__ G) {
    __shared__ __align__(16) _Float16 EkA[32 * 64];  // [k][m 64]
    __shared__ __align__(16) _Float16 EkB[32 * 64];  // [k][n 64]
    int tid = threadIdx.x;
    int lane = tid & 31;
    int wave = tid >> 5;
    int bz = blockIdx.z;
    int m_base = blockIdx.y * 64;
    int n_base = blockIdx.x * 64;
    const _Float16* Eb = Eh + (long)bz * 512 * 512;
    int wm = wave >> 1, wn = wave & 1;
    v8f acc00 = {}, acc01 = {}, acc10 = {}, acc11 = {};
    for (int kt = 0; kt < 16; ++kt) {
        int k0 = kt << 5;
#pragma unroll
        for (int j = 0; j < 2; ++j) {
            int seg = tid + (j << 7);          // 0..255: k = seg>>3, 8-half part
            int k = seg >> 3, part = (seg & 7) << 3;
            const _Float16* srcA = Eb + (long)(k0 + k) * 512 + m_base + part;
            const _Float16* srcB = Eb + (long)(k0 + k) * 512 + n_base + part;
            _Float16* dstA = EkA + k * 64 + part;
            _Float16* dstB = EkB + k * 64 + part;
#if HAVE_ASYNC_LDS
            __builtin_amdgcn_global_load_async_to_lds_b128(
                (gvec_ptr)srcA, (lvec_ptr)dstA, 0, 0);
            __builtin_amdgcn_global_load_async_to_lds_b128(
                (gvec_ptr)srcB, (lvec_ptr)dstB, 0, 0);
#else
            *(float4*)dstA = *(const float4*)srcA;
            *(float4*)dstB = *(const float4*)srcB;
#endif
        }
#if HAVE_ASYNC_LDS
        __builtin_amdgcn_s_wait_asynccnt(0);
#endif
        __syncthreads();
        v16h a0 = frag_ld_a_kmaj(EkA, wm * 32);
        v16h a1 = frag_ld_a_kmaj(EkA, wm * 32 + 16);
        v16h b0 = frag_ld_b_kmaj(EkB, wn * 32);
        v16h b1 = frag_ld_b_kmaj(EkB, wn * 32 + 16);
        acc00 = WMMA_F16(a0, b0, acc00);
        acc01 = WMMA_F16(a0, b1, acc01);
        acc10 = WMMA_F16(a1, b0, acc10);
        acc11 = WMMA_F16(a1, b1, acc11);
        __syncthreads();
    }
    float* Gb = G + (long)bz * 512 * 512;
    int rowoff = (lane & 16) ? 8 : 0;
    int col_l = lane & 15;
    v8f* accs[2][2] = {{&acc00, &acc01}, {&acc10, &acc11}};
    for (int mi = 0; mi < 2; ++mi)
        for (int ni = 0; ni < 2; ++ni) {
            int gcol = n_base + wn * 32 + ni * 16 + col_l;
#pragma unroll
            for (int r = 0; r < 8; ++r) {
                long grow = m_base + wm * 32 + mi * 16 + r + rowoff;
                Gb[grow * 512 + gcol] = (*accs[mi][ni])[r];
            }
        }
}

__global__ __launch_bounds__(256)
void cvt_f32_f16(const float* __restrict__ in, _Float16* __restrict__ out) {
    int i = blockIdx.x * 256 + threadIdx.x;
    out[i] = (_Float16)in[i];
}

// ---------------------------------------------------------------------------
// Fused attention: per (row-tile 16, head, batch), one wave.
// scores = clip(Q K^T / sqrt(8), -5, 5); softmax = exp/sum (clip bounds exp,
// so no running max needed); out = P V. dk=8 zero-padded to K=32 for WMMA.
// ---------------------------------------------------------------------------
__global__ __launch_bounds__(32)
void attn_kernel(const float* __restrict__ q, const float* __restrict__ k,
                 const float* __restrict__ v, float* __restrict__ o) {
    __shared__ __align__(16) _Float16 Qs[16 * 32];
    __shared__ __align__(16) _Float16 Ks[32 * 32];  // n-major: [32 n][32 dpad]
    __shared__ __align__(16) _Float16 Ps[16 * 32];
    __shared__ __align__(16) _Float16 Vs[16 * 32];  // n-major: [16 dpad][32 kk]
    int lane = threadIdx.x;
    int b = blockIdx.z, h = blockIdx.y, m0 = blockIdx.x * 16;
    const float scale = 0.35355339059327373f;       // 8^-0.5
    long base = (long)b * 512 * 64 + h * 8;
#pragma unroll
    for (int i = 0; i < 16; ++i) {
        int idx = lane + (i << 5);
        int r = idx >> 5, c = idx & 31;
        float val = (c < 8) ? q[base + (long)(m0 + r) * 64 + c] * scale : 0.f;
        Qs[r * 32 + c] = (_Float16)val;
    }
    v16h aq = frag_ld_a(Qs);
    v8f acc = {};
    float rs[8] = {0, 0, 0, 0, 0, 0, 0, 0};
    int colloc = lane & 15;
    int rowoff = (lane & 16) ? 8 : 0;
    for (int t = 0; t < 16; ++t) {
        int c0 = t << 5;
        int n = lane;
#pragma unroll
        for (int d = 0; d < 32; ++d) {
            float val = (d < 8) ? k[base + (long)(c0 + n) * 64 + d] : 0.f;
            Ks[n * 32 + d] = (_Float16)val;
        }
        v16h bk0 = frag_ld_b(Ks);
        v16h bk1 = frag_ld_b(Ks + 16 * 32);
        v8f z = {};
        v8f s0 = WMMA_F16(aq, bk0, z);
        v8f s1 = WMMA_F16(aq, bk1, z);
#pragma unroll
        for (int r = 0; r < 8; ++r) {
            float p0 = __expf(fminf(fmaxf(s0[r], -5.f), 5.f));
            float p1 = __expf(fminf(fmaxf(s1[r], -5.f), 5.f));
            rs[r] += p0 + p1;
            Ps[(r + rowoff) * 32 + colloc] = (_Float16)p0;
            Ps[(r + rowoff) * 32 + 16 + colloc] = (_Float16)p1;
        }
        v16h ap = frag_ld_a(Ps);
#pragma unroll
        for (int i = 0; i < 16; ++i) {
            int idx = lane + (i << 5);
            int d = idx >> 5, kk = idx & 31;
            float val = (d < 8) ? v[base + (long)(c0 + kk) * 64 + d] : 0.f;
            Vs[d * 32 + kk] = (_Float16)val;
        }
        v16h bv = frag_ld_b(Vs);
        acc = WMMA_F16(ap, bv, acc);
    }
#pragma unroll
    for (int r = 0; r < 8; ++r) {
        rs[r] += __shfl_xor(rs[r], 1, 32);
        rs[r] += __shfl_xor(rs[r], 2, 32);
        rs[r] += __shfl_xor(rs[r], 4, 32);
        rs[r] += __shfl_xor(rs[r], 8, 32);
    }
    if (colloc < 8) {
#pragma unroll
        for (int r = 0; r < 8; ++r)
            o[base + (long)(m0 + r + rowoff) * 64 + colloc] = acc[r] / rs[r];
    }
}

// ---------------------------------------------------------------------------
// Subspace iteration on G = E^T E (one workgroup per batch): top-8 eigvecs via
// power iteration + modified Gram-Schmidt, keep top-5 by Rayleigh quotient.
// ---------------------------------------------------------------------------
__device__ __forceinline__ float block_reduce256(float val, float* red) {
    int t = threadIdx.x;
    red[t] = val;
    __syncthreads();
    for (int s = 128; s > 0; s >>= 1) {
        if (t < s) red[t] += red[t + s];
        __syncthreads();
    }
    float r = red[0];
    __syncthreads();
    return r;
}

__global__ __launch_bounds__(256)
void svd_subspace(const float* __restrict__ G, float* __restrict__ Vsel,
                  float* __restrict__ sig) {
    __shared__ float X[512 * 8];
    __shared__ float Y[512 * 8];
    __shared__ float red[256];
    __shared__ float lam[8];
    __shared__ int ord[8];
    int b = blockIdx.x;
    const float* Gb = G + (long)b * 512 * 512;
    int t = threadIdx.x;
    for (int r = t; r < 512; r += 256)
        for (int c = 0; c < 8; ++c) {
            unsigned hh = (unsigned)(r * 8 + c + 1) * 2654435761u;
            X[r * 8 + c] = ((hh >> 8) & 0xFFFF) * (1.f / 65536.f) - 0.5f;
        }
    __syncthreads();
    for (int it = 0; it < 13; ++it) {
        for (int r = t; r < 512; r += 256) {
            float a[8] = {0, 0, 0, 0, 0, 0, 0, 0};
            for (int k2 = 0; k2 < 512; ++k2) {
                float g = Gb[(long)r * 512 + k2];
#pragma unroll
                for (int c = 0; c < 8; ++c) a[c] += g * X[k2 * 8 + c];
            }
#pragma unroll
            for (int c = 0; c < 8; ++c) Y[r * 8 + c] = a[c];
        }
        __syncthreads();
        if (it == 12) break;  // keep last multiply in Y for Rayleigh quotients
        for (int c = 0; c < 8; ++c) {
            for (int p = 0; p < c; ++p) {
                float part = 0.f;
                for (int r = t; r < 512; r += 256) part += X[r * 8 + p] * Y[r * 8 + c];
                float d = block_reduce256(part, red);
                for (int r = t; r < 512; r += 256) Y[r * 8 + c] -= d * X[r * 8 + p];
                __syncthreads();
            }
            float part = 0.f;
            for (int r = t; r < 512; r += 256) part += Y[r * 8 + c] * Y[r * 8 + c];
            float nrm = sqrtf(block_reduce256(part, red)) + 1e-20f;
            for (int r = t; r < 512; r += 256) X[r * 8 + c] = Y[r * 8 + c] / nrm;
            __syncthreads();
        }
    }
    for (int c = 0; c < 8; ++c) {
        float part = 0.f;
        for (int r = t; r < 512; r += 256) part += X[r * 8 + c] * Y[r * 8 + c];
        float l = block_reduce256(part, red);
        if (t == 0) lam[c] = l;
    }
    __syncthreads();
    if (t == 0) {
        bool used[8] = {};
        for (int j = 0; j < 5; ++j) {
            int best = 0;
            float bv = -1e30f;
            for (int c = 0; c < 8; ++c)
                if (!used[c] && lam[c] > bv) { bv = lam[c]; best = c; }
            used[best] = true;
            ord[j] = best;
            sig[b * 5 + j] = sqrtf(fmaxf(bv, 1e-12f));  // sigma = sqrt(lambda)
        }
    }
    __syncthreads();
    for (int j = 0; j < 5; ++j) {
        int c = ord[j];
        for (int r = t; r < 512; r += 256)
            Vsel[((long)b * 512 + r) * 5 + j] = X[r * 8 + c];
    }
}

// pos[b,i,j] = (E v_j)_i / sqrt(sigma_j);  pos[b,i,5+j] = v_j[i]*sqrt(sigma_j)
__global__ __launch_bounds__(160)
void pos_kernel(const float* __restrict__ E, const float* __restrict__ Vsel,
                const float* __restrict__ sig, float* __restrict__ pos) {
    int bi = blockIdx.x;
    int b = bi >> 9, i = bi & 511;
    int w = threadIdx.x >> 5, lane = threadIdx.x & 31;   // w = column j (0..4)
    const float* Er = E + ((long)b * 512 + i) * 512;
    const float* Vb = Vsel + (long)b * 512 * 5;
    float dv = 0.f;
    for (int k2 = lane; k2 < 512; k2 += 32) dv += Er[k2] * Vb[k2 * 5 + w];
    for (int m = 1; m < 32; m <<= 1) dv += __shfl_xor(dv, m, 32);
    if (lane == 0) {
        float s = sig[b * 5 + w];
        float ssq = sqrtf(s);
        float* pr = pos + (long)bi * 10;
        pr[w] = dv / ssq;
        pr[5 + w] = Vb[i * 5 + w] * ssq;
    }
}

// x128[b,i,j] = sum_o relu(nf*W1[0,j] + oid*W1[1,j] + b1[j])
__global__ __launch_bounds__(128)
void token_x(const float* __restrict__ n_feat, const float* __restrict__ w1,
             const float* __restrict__ b1, float* __restrict__ x128) {
    int bi = blockIdx.x;
    int j = threadIdx.x;
    int b = bi >> 9, i = bi & 511;
    float acc = 0.f;
    for (int o = 0; o < 3; ++o) {
        float nf = n_feat[((long)b * 3 + o) * 512 + i];
        float oid = (o + 1) * 0.1f;
        float vv = nf * w1[j] + oid * w1[128 + j] + b1[j];
        acc += fmaxf(vv, 0.f);
    }
    x128[(long)bi * 128 + j] = acc;
}

__global__ __launch_bounds__(64)
void pos_add(const float* __restrict__ pos, const float* __restrict__ wp,
             const float* __restrict__ bp, float* __restrict__ n) {
    int r = blockIdx.x;
    int d = threadIdx.x;
    float acc = bp[d];
    const float* pr = pos + (long)r * 10;
#pragma unroll
    for (int t = 0; t < 10; ++t) acc += pr[t] * wp[t * 64 + d];
    n[(long)r * 64 + d] += acc;
}

__global__ __launch_bounds__(32)
void ln_kernel(const float* __restrict__ in, float* __restrict__ out,
               const float* __restrict__ g, const float* __restrict__ beta) {
    int r = blockIdx.x;
    int lane = threadIdx.x;
    const float* x = in + (long)r * 64;
    float x0 = x[lane], x1 = x[lane + 32];
    float s = x0 + x1;
    for (int m = 1; m < 32; m <<= 1) s += __shfl_xor(s, m, 32);
    float mean = s * (1.f / 64.f);
    float d0 = x0 - mean, d1 = x1 - mean;
    float vv = d0 * d0 + d1 * d1;
    for (int m = 1; m < 32; m <<= 1) vv += __shfl_xor(vv, m, 32);
    float rstd = rsqrtf(vv * (1.f / 64.f) + 1e-5f);
    out[(long)r * 64 + lane] = d0 * rstd * g[lane] + beta[lane];
    out[(long)r * 64 + lane + 32] = d1 * rstd * g[lane + 32] + beta[lane + 32];
}

__global__ __launch_bounds__(64)
void gsum_kernel(const float* __restrict__ nfin, float* __restrict__ gsum) {
    int b = blockIdx.x;
    int d = threadIdx.x;
    float s = 0.f;
    for (int i = 0; i < 512; ++i) s += nfin[((long)b * 512 + i) * 64 + d];
    gsum[b * 64 + d] = s;
}

__global__ __launch_bounds__(128)
void small_mlp(const float* __restrict__ in, const float* __restrict__ w1,
               const float* __restrict__ b1, const float* __restrict__ w2,
               const float* __restrict__ b2, float* __restrict__ out) {
    __shared__ float hbuf[128];
    int b = blockIdx.x;
    int j = threadIdx.x;
    float acc = b1[j];
    for (int k2 = 0; k2 < 64; ++k2) acc += in[b * 64 + k2] * w1[k2 * 128 + j];
    hbuf[j] = fmaxf(acc, 0.f);
    __syncthreads();
    if (j < 64) {
        float a2 = b2[j];
        for (int k2 = 0; k2 < 128; ++k2) a2 += hbuf[k2] * w2[k2 * 64 + j];
        out[b * 64 + j] = fmaxf(a2, 0.f);
    }
}

// state_in[s,d] = sum_l n_pad[pids[pids[s]], indices[s,l], d]  (row 512 == 0)
__global__ __launch_bounds__(64)
void state_gather(const float* __restrict__ nfin, const int* __restrict__ pids,
                  const int* __restrict__ indices, float* __restrict__ state_in) {
    int s = blockIdx.x;
    int d = threadIdx.x;
    int pid2 = pids[pids[s]];
    float acc = 0.f;
    for (int l = 0; l < 64; ++l) {
        int idx = indices[s * 64 + l];
        if (idx < 512) acc += nfin[((long)pid2 * 512 + idx) * 64 + d];
    }
    state_in[(long)s * 64 + d] = acc;
}

__global__ __launch_bounds__(256)
void final_kernel(const float* __restrict__ state_out, const float* __restrict__ inst,
                  const float* __restrict__ ltab, const float* __restrict__ nfin,
                  const int* __restrict__ pids, const int* __restrict__ lids,
                  const int* __restrict__ vids, const float* __restrict__ wpred,
                  const float* __restrict__ bpred, float* __restrict__ out) {
    int s = blockIdx.x * 256 + threadIdx.x;
    int pid = pids[s], vid = vids[s], lid = lids[s];
    float o0 = bpred[0], o1 = bpred[1];
    for (int d = 0; d < 64; ++d) {
        float lv = (vid < 512) ? nfin[((long)pid * 512 + vid) * 64 + d] : 0.f;
        float val = state_out[(long)s * 64 + d] + inst[pid * 64 + d] + ltab[lid * 64 + d] + lv;
        o0 += val * wpred[d * 2 + 0];
        o1 += val * wpred[d * 2 + 1];
    }
    out[s * 2 + 0] = o0;
    out[s * 2 + 1] = o1;
}

// ---------------------------------------------------------------------------
// Host launcher. d_in: top-level setup_inputs order; params pytree-flattened
// (dict keys sorted alphabetically at each level):
//  0 n_feat  1 e_feat  2 pids_index  3 lids  4 vids  5 indices
//  6..9 agg{b1 b2 w1 w2}
//  10..29 block0: b_e1 b_e2 b_n1 b_n2 g_e1 g_e2 g_n1 g_n2
//                 mlp_e{b1 b2 w1 w2} mlp_n{b1 b2 w1 w2} wk wo wq wv
//  30..49 block1  50 bpred  51..54 graph{b1 b2 w1 w2}  55 layer_table
//  56..62 token{b1 b2 bp edge_table w1 w2 wp}  63 wpred
// ---------------------------------------------------------------------------
extern "C" void kernel_launch(void* const* d_in, const int* in_sizes, int n_in,
                              void* d_out, int out_size, void* d_ws, size_t ws_size,
                              hipStream_t stream) {
    (void)in_sizes; (void)n_in; (void)out_size; (void)ws_size;
    const float* n_feat = (const float*)d_in[0];
    const float* e_feat = (const float*)d_in[1];
    const int* pids = (const int*)d_in[2];
    const int* lids = (const int*)d_in[3];
    const int* vids = (const int*)d_in[4];
    const int* indices = (const int*)d_in[5];
    const float* agg_b1 = (const float*)d_in[6];
    const float* agg_b2 = (const float*)d_in[7];
    const float* agg_w1 = (const float*)d_in[8];
    const float* agg_w2 = (const float*)d_in[9];
    const float* bpred = (const float*)d_in[50];
    const float* graph_b1 = (const float*)d_in[51];
    const float* graph_b2 = (const float*)d_in[52];
    const float* graph_w1 = (const float*)d_in[53];
    const float* graph_w2 = (const float*)d_in[54];
    const float* ltab = (const float*)d_in[55];
    const float* tok_b1 = (const float*)d_in[56];
    const float* tok_b2 = (const float*)d_in[57];
    const float* tok_bp = (const float*)d_in[58];
    const float* tok_w1 = (const float*)d_in[60];
    const float* tok_w2 = (const float*)d_in[61];
    const float* tok_wp = (const float*)d_in[62];
    const float* wpred = (const float*)d_in[63];

    float* ws = (float*)d_ws;
    size_t off = 0;
    auto take = [&](size_t cnt) { float* p = ws + off; off += cnt; return p; };
    float* G = take(2l * 512 * 512);
    _Float16* Ehalf = (_Float16*)take(2l * 512 * 512 / 2);
    float* Vsel = take(2l * 512 * 5);
    float* sig = take(16);
    float* pos = take(2l * 512 * 10);
    float* x128 = take(2l * 512 * 128);   // also reused as MLP hidden buffer
    float* n = take(2l * 512 * 64);
    float* nl = take(2l * 512 * 64);
    float* n2 = take(2l * 512 * 64);
    float* qb = take(2l * 512 * 64);
    float* kb = take(2l * 512 * 64);
    float* vb = take(2l * 512 * 64);
    float* attnO = take(2l * 512 * 64);
    float* gsum = take(128);
    float* inst = take(128);
    float* state_in = take(2048l * 64);
    float* state_h = take(2048l * 128);
    float* state_out = take(2048l * 64);

    auto gemm = [&](const float* A, int lda, const float* Bm, int ldb,
                    float* C, int ldc, const float* bias, const float* resid,
                    int relu, int M, int N, int K) {
        dim3 grid(N / 64, M / 64, 1);
        wmma_gemm_rr<<<grid, dim3(128), 0, stream>>>(A, lda, Bm, ldb, C, ldc,
                                                     bias, resid, relu, K);
    };

    // --- SVD positional features ---
    cvt_f32_f16<<<dim3(2048), dim3(256), 0, stream>>>(e_feat, Ehalf);
    wmma_gram<<<dim3(8, 8, 2), dim3(128), 0, stream>>>(Ehalf, G);
    svd_subspace<<<dim3(2), dim3(256), 0, stream>>>(G, Vsel, sig);
    pos_kernel<<<dim3(1024), dim3(160), 0, stream>>>(e_feat, Vsel, sig, pos);

    // --- token embedding ---
    token_x<<<dim3(1024), dim3(128), 0, stream>>>(n_feat, tok_w1, tok_b1, x128);
    gemm(x128, 128, tok_w2, 64, n, 64, tok_b2, nullptr, 1, 1024, 64, 128);
    pos_add<<<dim3(1024), dim3(64), 0, stream>>>(pos, tok_wp, tok_bp, n);

    // --- encoder blocks (edge path is dead w.r.t. output: skipped) ---
    for (int blk = 0; blk < 2; ++blk) {
        int base = 10 + blk * 20;
        const float* b_n1 = (const float*)d_in[base + 2];
        const float* b_n2 = (const float*)d_in[base + 3];
        const float* g_n1 = (const float*)d_in[base + 6];
        const float* g_n2 = (const float*)d_in[base + 7];
        const float* mn_b1 = (const float*)d_in[base + 12];
        const float* mn_b2 = (const float*)d_in[base + 13];
        const float* mn_w1 = (const float*)d_in[base + 14];
        const float* mn_w2 = (const float*)d_in[base + 15];
        const float* wk = (const float*)d_in[base + 16];
        const float* wo = (const float*)d_in[base + 17];
        const float* wq = (const float*)d_in[base + 18];
        const float* wv = (const float*)d_in[base + 19];

        ln_kernel<<<dim3(1024), dim3(32), 0, stream>>>(n, nl, g_n1, b_n1);
        gemm(nl, 64, wq, 64, qb, 64, nullptr, nullptr, 0, 1024, 64, 64);
        gemm(nl, 64, wk, 64, kb, 64, nullptr, nullptr, 0, 1024, 64, 64);
        gemm(nl, 64, wv, 64, vb, 64, nullptr, nullptr, 0, 1024, 64, 64);
        attn_kernel<<<dim3(32, 8, 2), dim3(32), 0, stream>>>(qb, kb, vb, attnO);
        gemm(attnO, 64, wo, 64, n, 64, nullptr, nl, 0, 1024, 64, 64);
        ln_kernel<<<dim3(1024), dim3(32), 0, stream>>>(n, n2, g_n2, b_n2);
        gemm(n2, 64, mn_w1, 128, x128, 128, mn_b1, nullptr, 1, 1024, 128, 64);
        gemm(x128, 128, mn_w2, 64, n, 64, mn_b2, n2, 1, 1024, 64, 128);
    }

    // --- heads ---
    gsum_kernel<<<dim3(2), dim3(64), 0, stream>>>(n, gsum);
    small_mlp<<<dim3(2), dim3(128), 0, stream>>>(gsum, graph_w1, graph_b1, graph_w2,
                                                 graph_b2, inst);
    state_gather<<<dim3(2048), dim3(64), 0, stream>>>(n, pids, indices, state_in);
    gemm(state_in, 64, agg_w1, 128, state_h, 128, agg_b1, nullptr, 1, 2048, 128, 64);
    gemm(state_h, 128, agg_w2, 64, state_out, 64, agg_b2, nullptr, 1, 2048, 64, 128);
    final_kernel<<<dim3(8), dim3(256), 0, stream>>>(state_out, inst, ltab, n, pids,
                                                    lids, vids, wpred, bpred,
                                                    (float*)d_out);
}
